// MultiHeadSelfAttention_59931973648640
// MI455X (gfx1250) — matrix operand order repaired and checked by reading, not compiled, not run on hardware
//
#include <hip/hip_runtime.h>
#include <hip/hip_bf16.h>

// MultiHeadSelfAttention for MI455X (gfx1250, wave32, WMMA bf16 path).
//
// Pipeline:
//   1) fp32 -> bf16 conversion of x and the four weight matrices (workspace)
//   2) fused Q/K/V projection GEMM (bf16 WMMA 16x16x32, fp32 accum),
//      double-buffered LDS + register prefetch, outputs [B,H,S,D] bf16
//   3) flash-attention kernel (online softmax, WMMA for QK^T and P*V,
//      K/V tile register prefetch), context -> [B,S,E] bf16
//   4) output projection GEMM -> d_out fp32
//
// Workspace layout (u16 elements):
//   xb   [8192*1024]            @ 0
//   Wqb/Wkb/Wvb/Wob [1024*1024] each
//   Qb/Kb/Vb [B,H,S,D]=8388608 each
//   ctx  [B,S,E]=8388608
// Total = 92,274,688 bytes (~88 MB).

typedef unsigned short u16;
typedef unsigned int u32;

typedef __attribute__((ext_vector_type(8)))  __bf16 vbf8;
typedef __attribute__((ext_vector_type(16))) __bf16 vbf16;
typedef __attribute__((ext_vector_type(8)))  float  vf8;

static constexpr int E = 1024;
static constexpr int S = 2048;
static constexpr int H = 16;
static constexpr int D = 64;
static constexpr int BATCH = 4;
static constexpr int M_TOK = BATCH * S; // 8192

__device__ __forceinline__ u16 f2bf(float f) {
  union { float f; u32 u; } v;
  v.f = f;
  u32 r = ((v.u >> 16) & 1u) + 0x7FFFu; // round-to-nearest-even
  return (u16)((v.u + r) >> 16);
}

__device__ __forceinline__ vbf16 cat8(vbf8 lo, vbf8 hi) {
  vbf16 r;
#pragma unroll
  for (int i = 0; i < 8; ++i) { r[i] = lo[i]; r[i + 8] = hi[i]; }
  return r;
}

// A-fragment (16x32 bf16, MxK) per CDNA5 layout:
// lane<16: row=lane, v0..3 = k0..7, v4..7 = k16..23 ; lane>=16: +8 in k.
__device__ __forceinline__ vbf16 load_a_frag(const u16* base, int ld) {
  int lane = threadIdx.x & 31;
  const u16* p = base + (lane & 15) * ld + ((lane >> 4) << 3);
  vbf8 lo = *(const vbf8*)(p);
  vbf8 hi = *(const vbf8*)(p + 16);
  return cat8(lo, hi);
}

// B-fragment (32x16 bf16, KxN) from row-major [N][K] tile:
// lane<16: n=lane, v0..3 = k0..7, v4..7 = k16..23 ; lane>=16: n=lane-16, +8 in k.
__device__ __forceinline__ vbf16 load_b_frag(const u16* base, int ld) {
  int lane = threadIdx.x & 31;
  const u16* p = base + (lane & 15) * ld + ((lane >> 4) << 3);
  vbf8 lo = *(const vbf8*)(p);
  vbf8 hi = *(const vbf8*)(p + 16);
  return cat8(lo, hi);
}

__device__ __forceinline__ vf8 wmma_bf16(vbf16 a, vbf16 b, vf8 c) {
  return __builtin_amdgcn_wmma_f32_16x16x32_bf16(
      /*neg_a=*/false, a, /*neg_b=*/false, b,
      /*c_mod=*/(short)0, c, /*reuse_a=*/false, /*reuse_b=*/false);
}

// ---------------------------------------------------------------------------
// fp32 -> bf16 conversion, 4 elements/thread
// ---------------------------------------------------------------------------
__global__ void cvt_f32_bf16(const float* __restrict__ src, u16* __restrict__ dst, int n4) {
  int i = blockIdx.x * blockDim.x + threadIdx.x;
  if (i >= n4) return;
  float4 f = ((const float4*)src)[i];
  u32 lo = (u32)f2bf(f.x) | ((u32)f2bf(f.y) << 16);
  u32 hi = (u32)f2bf(f.z) | ((u32)f2bf(f.w) << 16);
  ((uint2*)dst)[i] = make_uint2(lo, hi);
}

// ---------------------------------------------------------------------------
// GEMM: C[M,N] = A[M,K] * W[N,K]^T + bias   (M=8192, N=K=1024)
// mode==1: z in {0,1,2} selects (W,b,out); out is bf16 in [B,H,S,D] layout.
// mode==0: fp32 output to outF[M,N] (the final projection).
// Block: 256 threads = 8 waves (4 M x 2 N), block tile 128x128, K-step 32.
// Double-buffered LDS ping-pong; next tile's global loads issued right after
// the staging barrier so they overlap fragment loads + 8 WMMAs.
// ---------------------------------------------------------------------------
__global__ __launch_bounds__(256) void gemm_kernel(
    const u16* __restrict__ A,
    const u16* __restrict__ W0, const u16* __restrict__ W1, const u16* __restrict__ W2,
    const float* __restrict__ b0, const float* __restrict__ b1, const float* __restrict__ b2,
    u16* __restrict__ o0, u16* __restrict__ o1, u16* __restrict__ o2,
    float* __restrict__ outF, int mode) {
  constexpr int LDT = 40; // 32 + 8 pad (u16) -> 80B row stride, conflict-free b128 reads
  __shared__ u16 lA[2][128][LDT];
  __shared__ u16 lB[2][128][LDT];

  int z = blockIdx.z;
  const u16* W = (z == 0) ? W0 : (z == 1 ? W1 : W2);
  const float* bias = (z == 0) ? b0 : (z == 1 ? b1 : b2);
  u16* outq = (z == 0) ? o0 : (z == 1 ? o1 : o2);

  int m0 = blockIdx.y * 128;
  int n0 = blockIdx.x * 128;
  int tid = threadIdx.x, lane = tid & 31, wave = tid >> 5;
  int wm = wave & 3, wn = wave >> 2;

  vf8 vz = {0.f, 0.f, 0.f, 0.f, 0.f, 0.f, 0.f, 0.f};
  vf8 acc[2][4];
#pragma unroll
  for (int i = 0; i < 2; ++i)
#pragma unroll
    for (int j = 0; j < 4; ++j) acc[i][j] = vz;

  int row = tid >> 1, half = tid & 1;
  const u16* gA = A + (size_t)(m0 + row) * E + half * 16;
  const u16* gW = W + (size_t)(n0 + row) * E + half * 16;

  // prefetch first tile into registers
  uint4 pa0 = ((const uint4*)gA)[0], pa1 = ((const uint4*)gA)[1];
  uint4 pw0 = ((const uint4*)gW)[0], pw1 = ((const uint4*)gW)[1];

  for (int k0 = 0; k0 < E; k0 += 32) {
    int buf = (k0 >> 5) & 1;
    { // commit staged registers to this iteration's LDS buffer
      uint4* d = (uint4*)&lA[buf][row][half * 16];
      d[0] = pa0; d[1] = pa1;
      uint4* e = (uint4*)&lB[buf][row][half * 16];
      e[0] = pw0; e[1] = pw1;
    }
    __syncthreads();

    if (k0 + 32 < E) { // issue next tile's global loads; in flight during WMMAs
      const uint4* sA = (const uint4*)(gA + k0 + 32);
      const uint4* sW = (const uint4*)(gW + k0 + 32);
      pa0 = sA[0]; pa1 = sA[1];
      pw0 = sW[0]; pw1 = sW[1];
    }

    vbf16 af[2], bfm[4];
#pragma unroll
    for (int tm = 0; tm < 2; ++tm) af[tm] = load_a_frag(&lA[buf][wm * 32 + tm * 16][0], LDT);
#pragma unroll
    for (int tn = 0; tn < 4; ++tn) bfm[tn] = load_b_frag(&lB[buf][wn * 64 + tn * 16][0], LDT);
#pragma unroll
    for (int tm = 0; tm < 2; ++tm)
#pragma unroll
      for (int tn = 0; tn < 4; ++tn)
        acc[tm][tn] = wmma_bf16(af[tm], bfm[tn], acc[tm][tn]);
    // no trailing barrier needed: reads of buf are consumed by WMMA (dscnt wait)
    // before this thread's next-iteration store, and the next iteration's
    // staging barrier orders buffer reuse two iterations out.
  }

  // Epilogue: C layout -> vgpr r holds (M = base + r (+8 for upper half-lanes), N = lane&15)
#pragma unroll
  for (int tm = 0; tm < 2; ++tm) {
#pragma unroll
    for (int tn = 0; tn < 4; ++tn) {
      int nglob = n0 + wn * 64 + tn * 16 + (lane & 15);
      float bv = bias[nglob];
      int mbase = m0 + wm * 32 + tm * 16 + ((lane >> 4) << 3);
#pragma unroll
      for (int r = 0; r < 8; ++r) {
        int mg = mbase + r;
        float val = acc[tm][tn][r] + bv;
        if (mode == 0) {
          outF[(size_t)mg * E + nglob] = val;
        } else {
          int bb = mg >> 11, ss = mg & (S - 1);
          int hh = nglob >> 6, dd = nglob & 63;
          outq[(((size_t)bb * H + hh) * S + ss) * D + dd] = f2bf(val);
        }
      }
    }
  }
}

// ---------------------------------------------------------------------------
// Flash attention: grid (S/128, H, B), 256 threads (8 waves: 4 M x 2 N).
// 128-query tile, 64-key tiles, online softmax, bf16 WMMA, fp32 stats/accum.
// Next K/V tile is prefetched into registers right after the staging barrier.
// ---------------------------------------------------------------------------
__global__ __launch_bounds__(256) void attn_kernel(
    const u16* __restrict__ Q, const u16* __restrict__ K, const u16* __restrict__ V,
    const int* __restrict__ mask, u16* __restrict__ ctx) {
  constexpr int LDQ = 72; // 64 + 8 pad
  __shared__ u16 sQ[128][LDQ];
  __shared__ u16 sK[64][LDQ];
  __shared__ u16 sVt[64][LDQ];  // V transposed: [d][key]
  __shared__ u16 sP[128][LDQ];  // P tile bf16: [query][key]
  __shared__ float rowM[128], rowL[128], rowA[128];
  __shared__ float stMax[2][128], stSum[2][128];

  int b = blockIdx.z, h = blockIdx.y;
  int q0 = blockIdx.x * 128;
  int tid = threadIdx.x, lane = tid & 31, wave = tid >> 5;
  int wm = wave & 3, wn = wave >> 2;
  size_t bh = ((size_t)(b * H + h)) * S * D;

  // load Q tile: 128 rows x 64 bf16 (8 uint4 per row)
  for (int idx = tid; idx < 128 * 8; idx += 256) {
    int r = idx >> 3, c = idx & 7;
    *(uint4*)&sQ[r][c * 8] = *(const uint4*)(Q + bh + (size_t)(q0 + r) * D + c * 8);
  }
  if (tid < 128) { rowM[tid] = -3.0e38f; rowL[tid] = 0.0f; }

  vf8 vz = {0.f, 0.f, 0.f, 0.f, 0.f, 0.f, 0.f, 0.f};
  vf8 oacc[2][2];
#pragma unroll
  for (int i = 0; i < 2; ++i)
#pragma unroll
    for (int j = 0; j < 2; ++j) oacc[i][j] = vz;

  // K/V staging indices: 64 rows x 8 uint4 = 512 chunks, 2 per thread
  int i0 = tid, i1 = tid + 256;
  int kr0r = i0 >> 3, kr0c = i0 & 7;
  int kr1r = i1 >> 3, kr1c = i1 & 7;

  // prefetch first K/V tile into registers
  uint4 kreg0 = *(const uint4*)(K + bh + (size_t)kr0r * D + kr0c * 8);
  uint4 kreg1 = *(const uint4*)(K + bh + (size_t)kr1r * D + kr1c * 8);
  uint4 vreg0 = *(const uint4*)(V + bh + (size_t)kr0r * D + kr0c * 8);
  uint4 vreg1 = *(const uint4*)(V + bh + (size_t)kr1r * D + kr1c * 8);

  __syncthreads(); // sQ + rowM/rowL visible

  for (int k0 = 0; k0 < S; k0 += 64) {
    // commit staged K tile and transposed V tile to LDS
    *(uint4*)&sK[kr0r][kr0c * 8] = kreg0;
    *(uint4*)&sK[kr1r][kr1c * 8] = kreg1;
    {
      union { uint4 q; u16 hh[8]; } t0, t1;
      t0.q = vreg0; t1.q = vreg1;
#pragma unroll
      for (int j = 0; j < 8; ++j) sVt[kr0c * 8 + j][kr0r] = t0.hh[j];
#pragma unroll
      for (int j = 0; j < 8; ++j) sVt[kr1c * 8 + j][kr1r] = t1.hh[j];
    }
    __syncthreads();

    if (k0 + 64 < S) { // prefetch next tile; in flight for the whole iteration
      size_t base = bh + (size_t)(k0 + 64) * D;
      kreg0 = *(const uint4*)(K + base + (size_t)kr0r * D + kr0c * 8);
      kreg1 = *(const uint4*)(K + base + (size_t)kr1r * D + kr1c * 8);
      vreg0 = *(const uint4*)(V + base + (size_t)kr0r * D + kr0c * 8);
      vreg1 = *(const uint4*)(V + base + (size_t)kr1r * D + kr1c * 8);
    }

    // scores: wave tile = 32 queries x 32 keys, contraction over d=64
    vf8 sc[2][2];
    sc[0][0] = sc[0][1] = sc[1][0] = sc[1][1] = vz;
#pragma unroll
    for (int kk = 0; kk < 64; kk += 32) {
      vbf16 aq[2], bk2[2];
#pragma unroll
      for (int tm = 0; tm < 2; ++tm) aq[tm] = load_a_frag(&sQ[wm * 32 + tm * 16][kk], LDQ);
#pragma unroll
      for (int tn = 0; tn < 2; ++tn) bk2[tn] = load_b_frag(&sK[wn * 32 + tn * 16][kk], LDQ);
#pragma unroll
      for (int tm = 0; tm < 2; ++tm)
#pragma unroll
        for (int tn = 0; tn < 2; ++tn)
          sc[tm][tn] = wmma_bf16(aq[tm], bk2[tn], sc[tm][tn]);
    }

    // scale + key mask + per-lane row max
    float pm[2][8];
#pragma unroll
    for (int tm = 0; tm < 2; ++tm)
#pragma unroll
      for (int r = 0; r < 8; ++r) pm[tm][r] = -3.0e38f;
#pragma unroll
    for (int tn = 0; tn < 2; ++tn) {
      int kcol = k0 + wn * 32 + tn * 16 + (lane & 15);
      float mb = (mask[b * S + kcol] == 0) ? -1.0e9f : 0.0f;
#pragma unroll
      for (int tm = 0; tm < 2; ++tm)
#pragma unroll
        for (int r = 0; r < 8; ++r) {
          float v = sc[tm][tn][r] * 0.125f + mb;
          sc[tm][tn][r] = v;
          pm[tm][r] = fmaxf(pm[tm][r], v);
        }
    }
    // reduce max across the 16 lanes sharing each row (halves stay disjoint)
#pragma unroll
    for (int tm = 0; tm < 2; ++tm)
#pragma unroll
      for (int r = 0; r < 8; ++r)
#pragma unroll
        for (int off = 1; off < 16; off <<= 1)
          pm[tm][r] = fmaxf(pm[tm][r], __shfl_xor(pm[tm][r], off));
    if ((lane & 15) == 0) {
      int rb = wm * 32 + ((lane >> 4) << 3);
#pragma unroll
      for (int tm = 0; tm < 2; ++tm)
#pragma unroll
        for (int r = 0; r < 8; ++r) stMax[wn][rb + tm * 16 + r] = pm[tm][r];
    }
    __syncthreads();
    if (tid < 128) {
      float mOld = rowM[tid];
      float mNew = fmaxf(mOld, fmaxf(stMax[0][tid], stMax[1][tid]));
      rowM[tid] = mNew;
      rowA[tid] = __expf(mOld - mNew);
    }
    __syncthreads();

    // P = exp(s - m), partial row sums, store P tile as bf16
    float ps[2][8];
#pragma unroll
    for (int tm = 0; tm < 2; ++tm)
#pragma unroll
      for (int r = 0; r < 8; ++r) ps[tm][r] = 0.0f;
#pragma unroll
    for (int tm = 0; tm < 2; ++tm) {
      int rb = wm * 32 + tm * 16 + ((lane >> 4) << 3);
#pragma unroll
      for (int tn = 0; tn < 2; ++tn) {
        int col = wn * 32 + tn * 16 + (lane & 15);
#pragma unroll
        for (int r = 0; r < 8; ++r) {
          float p = __expf(sc[tm][tn][r] - rowM[rb + r]);
          ps[tm][r] += p;
          sP[rb + r][col] = f2bf(p);
        }
      }
    }
#pragma unroll
    for (int tm = 0; tm < 2; ++tm)
#pragma unroll
      for (int r = 0; r < 8; ++r)
#pragma unroll
        for (int off = 1; off < 16; off <<= 1)
          ps[tm][r] += __shfl_xor(ps[tm][r], off);
    if ((lane & 15) == 0) {
      int rb = wm * 32 + ((lane >> 4) << 3);
#pragma unroll
      for (int tm = 0; tm < 2; ++tm)
#pragma unroll
        for (int r = 0; r < 8; ++r) stSum[wn][rb + tm * 16 + r] = ps[tm][r];
    }
    __syncthreads();
    if (tid < 128) rowL[tid] = rowL[tid] * rowA[tid] + stSum[0][tid] + stSum[1][tid];

    // rescale O accumulators, then O += P @ V (wave tile 32 queries x 32 d)
#pragma unroll
    for (int tm = 0; tm < 2; ++tm) {
      int rb = wm * 32 + tm * 16 + ((lane >> 4) << 3);
#pragma unroll
      for (int r = 0; r < 8; ++r) {
        float a = rowA[rb + r];
        oacc[tm][0][r] *= a;
        oacc[tm][1][r] *= a;
      }
    }
#pragma unroll
    for (int kk = 0; kk < 64; kk += 32) {
      vbf16 ap[2], bv2[2];
#pragma unroll
      for (int tm = 0; tm < 2; ++tm) ap[tm] = load_a_frag(&sP[wm * 32 + tm * 16][kk], LDQ);
#pragma unroll
      for (int tn = 0; tn < 2; ++tn) bv2[tn] = load_b_frag(&sVt[wn * 32 + tn * 16][kk], LDQ);
#pragma unroll
      for (int tm = 0; tm < 2; ++tm)
#pragma unroll
        for (int tn = 0; tn < 2; ++tn)
          oacc[tm][tn] = wmma_bf16(ap[tm], bv2[tn], oacc[tm][tn]);
    }
    __syncthreads(); // protects sK/sVt/sP reuse by the next iteration's staging
  }

  // normalize and store context (merged heads, [B,S,E] bf16)
#pragma unroll
  for (int tm = 0; tm < 2; ++tm) {
    int rb = wm * 32 + tm * 16 + ((lane >> 4) << 3);
#pragma unroll
    for (int tn = 0; tn < 2; ++tn) {
      int dcol = wn * 32 + tn * 16 + (lane & 15);
#pragma unroll
      for (int r = 0; r < 8; ++r) {
        float val = oacc[tm][tn][r] / rowL[rb + r];
        int sg = q0 + rb + r;
        ctx[((size_t)b * S + sg) * E + h * D + dcol] = f2bf(val);
      }
    }
  }
}

// ---------------------------------------------------------------------------
extern "C" void kernel_launch(void* const* d_in, const int* in_sizes, int n_in,
                              void* d_out, int out_size, void* d_ws, size_t ws_size,
                              hipStream_t stream) {
  const float* x  = (const float*)d_in[0];
  const int* mask = (const int*)d_in[1];
  const float* Wq = (const float*)d_in[2];
  const float* bq = (const float*)d_in[3];
  const float* Wk = (const float*)d_in[4];
  const float* bk = (const float*)d_in[5];
  const float* Wv = (const float*)d_in[6];
  const float* bv = (const float*)d_in[7];
  const float* Wo = (const float*)d_in[8];
  const float* bo = (const float*)d_in[9];

  u16* ws  = (u16*)d_ws;
  u16* xb  = ws;                      // 8192*1024
  u16* Wqb = xb  + (size_t)8388608;   // 1024*1024 each
  u16* Wkb = Wqb + (size_t)1048576;
  u16* Wvb = Wkb + (size_t)1048576;
  u16* Wob = Wvb + (size_t)1048576;
  u16* Qb  = Wob + (size_t)1048576;   // [B,H,S,D] each
  u16* Kb  = Qb  + (size_t)8388608;
  u16* Vb  = Kb  + (size_t)8388608;
  u16* Cb  = Vb  + (size_t)8388608;   // [B,S,E]

  cvt_f32_bf16<<<8192, 256, 0, stream>>>(x,  xb,  2097152);
  cvt_f32_bf16<<<1024, 256, 0, stream>>>(Wq, Wqb, 262144);
  cvt_f32_bf16<<<1024, 256, 0, stream>>>(Wk, Wkb, 262144);
  cvt_f32_bf16<<<1024, 256, 0, stream>>>(Wv, Wvb, 262144);
  cvt_f32_bf16<<<1024, 256, 0, stream>>>(Wo, Wob, 262144);

  // Q/K/V projections fused over gridDim.z
  gemm_kernel<<<dim3(E / 128, M_TOK / 128, 3), 256, 0, stream>>>(
      xb, Wqb, Wkb, Wvb, bq, bk, bv, Qb, Kb, Vb, nullptr, 1);

  // flash attention
  attn_kernel<<<dim3(S / 128, H, BATCH), 256, 0, stream>>>(Qb, Kb, Vb, mask, Cb);

  // output projection -> fp32 d_out
  gemm_kernel<<<dim3(E / 128, M_TOK / 128, 1), 256, 0, stream>>>(
      Cb, Wob, Wob, Wob, bo, bo, bo, nullptr, nullptr, nullptr, (float*)d_out, 0);
}
